// gmodule_841813590082
// MI455X (gfx1250) — compile-verified, matrix operand order checked by reading, not run on hardware
//
#include <hip/hip_runtime.h>
#include <hip/hip_bf16.h>
#include <math.h>

typedef __attribute__((ext_vector_type(16))) _Float16 v16h;
typedef __attribute__((ext_vector_type(8)))  float    v8f;

#define HID 64

__device__ __forceinline__ v8f wmma_f16(v16h a, v16h b, v8f c) {
  // D = A(16x32 f16) * B(32x16 f16) + C(16x16 f32)
  return __builtin_amdgcn_wmma_f32_16x16x32_f16(false, a, false, b, (short)0, c,
                                                false, false);
}

// Load W2 [64][64] row-major (k, n) into the WMMA B-register layout as f16.
// B[t][kh] covers output columns n = t*16 + (lane&15), K = kh*32 .. kh*32+31.
// 16-bit operand layout: lanes 0-15 hold K = {0..7, 16..23} (+kh*32),
// lanes 16-31 hold K = {8..15, 24..31} (+kh*32).
__device__ __forceinline__ void load_w2_B(const float* __restrict__ W2, v16h B[4][2]) {
  const int lane  = threadIdx.x & 31;
  const int n0    = lane & 15;
  const int kbase = (lane >> 4) * 8;
#pragma unroll
  for (int t = 0; t < 4; ++t)
#pragma unroll
    for (int kh = 0; kh < 2; ++kh)
#pragma unroll
      for (int i = 0; i < 16; ++i) {
        int k = kh * 32 + kbase + i + ((i >= 8) ? 8 : 0);
        B[t][kh][i] = (_Float16)W2[k * HID + t * 16 + n0];
      }
}

// One wave processes 16 rows per tile iteration:
//   x[DIN] -> relu(W1) -> [WMMA 64x64] relu -> dot(W3)+b3 -> OUT_MODE
// OUT_MODE 0: atomicAdd(out[sidx[row]], v)
// OUT_MODE 1: out[row] = v
// OUT_MODE 2: atomicAdd(out[sidx[row]], sigmoid(v))
template <int DIN, int OUT_MODE>
__global__ void __launch_bounds__(256, 1) mlp16_wmma_kernel(
    const float* __restrict__ in0, const int* __restrict__ g0,
    const float* __restrict__ in1, const int* __restrict__ g1,
    const float* __restrict__ in2,
    const float* __restrict__ W1, const float* __restrict__ b1,
    const float* __restrict__ W2, const float* __restrict__ b2,
    const float* __restrict__ W3, const float* __restrict__ b3,
    float* __restrict__ out, const int* __restrict__ sidx, int nRows) {
  v16h B[4][2];
  load_w2_B(W2, B);  // resident in VGPRs for the whole grid-stride loop

  const int lane  = threadIdx.x & 31;
  const int n0    = lane & 15;   // row (for A/out) and column (for B/C) id
  const int half  = lane >> 4;
  const int kbase = half * 8;

  const int wavesPerBlock = blockDim.x >> 5;
  const int wave   = blockIdx.x * wavesPerBlock + (threadIdx.x >> 5);
  const int nWaves = gridDim.x * wavesPerBlock;
  const int nTiles = (nRows + 15) >> 4;
  const float bias3 = b3[0];

  for (int tile = wave; tile < nTiles; tile += nWaves) {
    const int row = tile * 16 + n0;
    const int rc  = (row < nRows) ? row : 0;  // data mask, no divergence

    float x0 = in0[g0 ? g0[rc] : rc];
    float x1 = 0.f, x2 = 0.f;
    if (DIN > 1) x1 = in1[g1 ? g1[rc] : rc];
    if (DIN > 2) x2 = in2[rc];

    // ---- layer 1 (DIN -> 64) + relu, produced directly in A layout (f16)
    v16h a0, a1;
#pragma unroll
    for (int i = 0; i < 16; ++i) {
      const int k = kbase + i + ((i >= 8) ? 8 : 0);
#pragma unroll
      for (int kh = 0; kh < 2; ++kh) {
        const int kk = kh * 32 + k;
        float h = b1[kk];
        h = fmaf(x0, W1[kk], h);
        if (DIN > 1) h = fmaf(x1, W1[HID + kk], h);
        if (DIN > 2) h = fmaf(x2, W1[2 * HID + kk], h);
        h = fmaxf(h, 0.f);
        if (kh == 0) a0[i] = (_Float16)h; else a1[i] = (_Float16)h;
      }
    }

    // ---- layer 2 (64 -> 64) via WMMA (+bias via C init, relu),
    //      fused with layer-3 partial dot against W3
    float p[8];
#pragma unroll
    for (int r = 0; r < 8; ++r) p[r] = 0.f;
#pragma unroll
    for (int t = 0; t < 4; ++t) {
      v8f c;
      const float bb = b2[t * 16 + n0];
#pragma unroll
      for (int r = 0; r < 8; ++r) c[r] = bb;
      c = wmma_f16(a0, B[t][0], c);   // K = 0..31
      c = wmma_f16(a1, B[t][1], c);   // K = 32..63
      const float w3 = W3[t * 16 + n0];
#pragma unroll
      for (int r = 0; r < 8; ++r) p[r] = fmaf(fmaxf(c[r], 0.f), w3, p[r]);
    }

    // ---- reduce across the 16 lanes holding the columns of each row
    // (masks 8..1 stay inside the C-layout half-wave row groups)
#pragma unroll
    for (int mask = 8; mask >= 1; mask >>= 1)
#pragma unroll
      for (int r = 0; r < 8; ++r) p[r] += __shfl_xor(p[r], mask, 32);

    // lane n0<8 of each half emits row M = half*8 + n0
    if (n0 < 8) {
      const int orow = tile * 16 + half * 8 + n0;
      if (orow < nRows) {
        const float v = p[n0] + bias3;
        if (OUT_MODE == 0) {
          atomicAdd(&out[sidx[orow]], v);
        } else if (OUT_MODE == 1) {
          out[orow] = v;
        } else {
          atomicAdd(&out[sidx[orow]], 1.0f / (1.0f + __expf(-v)));
        }
      }
    }
  }
}

__global__ void __launch_bounds__(256) zero_kernel(float* __restrict__ p, int n) {
  int i = blockIdx.x * blockDim.x + threadIdx.x;
  if (i < n) p[i] = 0.f;
}

__global__ void __launch_bounds__(256) sigmoid_kernel(const float* __restrict__ in,
                                                      float* __restrict__ out, int n) {
  int i = blockIdx.x * blockDim.x + threadIdx.x;
  if (i < n) out[i] = 1.0f / (1.0f + __expf(-in[i]));
}

static inline int tile_blocks(int nRows) {
  int nTiles = (nRows + 15) / 16;
  int blocks = (nTiles + 7) / 8;  // 8 waves per 256-thread block
  return blocks > 2048 ? 2048 : (blocks < 1 ? 1 : blocks);
}

extern "C" void kernel_launch(void* const* d_in, const int* in_sizes, int n_in,
                              void* d_out, int out_size, void* d_ws, size_t ws_size,
                              hipStream_t stream) {
  (void)n_in; (void)ws_size;
  const float* nf  = (const float*)d_in[0];
  const float* ef  = (const float*)d_in[1];
  const int*   src = (const int*)d_in[2];
  const int*   dst = (const int*)d_in[3];
  const int*   gid = (const int*)d_in[4];
  const float* e_w1 = (const float*)d_in[5],  *e_b1 = (const float*)d_in[6];
  const float* e_w2 = (const float*)d_in[7],  *e_b2 = (const float*)d_in[8];
  const float* e_w3 = (const float*)d_in[9],  *e_b3 = (const float*)d_in[10];
  const float* n_w1 = (const float*)d_in[11], *n_b1 = (const float*)d_in[12];
  const float* n_w2 = (const float*)d_in[13], *n_b2 = (const float*)d_in[14];
  const float* n_w3 = (const float*)d_in[15], *n_b3 = (const float*)d_in[16];
  const float* m_w1 = (const float*)d_in[17], *m_b1 = (const float*)d_in[18];
  const float* m_w2 = (const float*)d_in[19], *m_b2 = (const float*)d_in[20];
  const float* m_w3 = (const float*)d_in[21], *m_b3 = (const float*)d_in[22];

  const int N = in_sizes[0];  // 50000 nodes
  const int E = in_sizes[1];  // 1600000 edges
  const int G = out_size;     // 512 graphs

  float* ws  = (float*)d_ws;
  float* agg = ws;            // [N] per-dst message sums
  float* rd  = ws + N;        // [G] readout accumulators
  float* unf = ws + N + G;    // [N] node-model output

  // zero agg + rd (poisoned between runs)
  const int nz = N + G;
  zero_kernel<<<(nz + 255) / 256, 256, 0, stream>>>(ws, nz);

  // edge MLP: [nf[src], nf[dst], ef] -> msg, scatter-add into agg[dst]
  mlp16_wmma_kernel<3, 0><<<tile_blocks(E), 256, 0, stream>>>(
      nf, src, nf, dst, ef, e_w1, e_b1, e_w2, e_b2, e_w3, e_b3, agg, dst, E);

  // node model: [nf, agg] -> unf
  mlp16_wmma_kernel<2, 1><<<tile_blocks(N), 256, 0, stream>>>(
      nf, nullptr, agg, nullptr, nullptr, n_w1, n_b1, n_w2, n_b2, n_w3, n_b3,
      unf, nullptr, N);

  // node-wise MLP with sigmoid, scatter-add into rd[graph_id]
  mlp16_wmma_kernel<1, 2><<<tile_blocks(N), 256, 0, stream>>>(
      unf, nullptr, nullptr, nullptr, nullptr, m_w1, m_b1, m_w2, m_b2, m_w3, m_b3,
      rd, gid, N);

  // final sigmoid over per-graph readout
  sigmoid_kernel<<<(G + 255) / 256, 256, 0, stream>>>(rd, (float*)d_out, G);
}
